// JointNetwork_23562190586541
// MI455X (gfx1250) — matrix operand order repaired and checked by reading, not compiled, not run on hardware
//
#include <hip/hip_runtime.h>
#include <hip/hip_bf16.h>

// ---- problem constants (match reference) ----
#define BB 8
#define TT 200
#define UU 100
#define JJ 512
#define VV 1024
#define TU (TT * UU)          // 20000 rows per batch
#define MTOT (BB * TU)        // 160000 rows total

typedef __attribute__((ext_vector_type(16))) _Float16 v16h;
typedef __attribute__((ext_vector_type(8)))  _Float16 v8h;
typedef __attribute__((ext_vector_type(8)))  float    v8f;

#define LDS_STRIDE 520   // 512 + 8 halves padding to break bank conflicts

// =====================================================================
// Pack a (nrows, 512) fp32 weight matrix into f16 WMMA B-fragments for
// v_wmma_f32_16x16x32_f16. B is 32(K)x16(N):
//   lane n  (n<16):  column N=n, vector elems 0..7 = K 0..7, 8..15 = K 16..23
//   lane n+16:       column N=n, vector elems 0..7 = K 8..15, 8..15 = K 24..31
// Packed order: [ks=0..15][ntile=0..nrows/16-1][lane=0..31][elem=0..15]
// (one fragment = 32 lanes x 32 B = 1 KB contiguous).
// =====================================================================
__global__ __launch_bounds__(256) void pack_w(
    const float* __restrict__ W, _Float16* __restrict__ Wp,
    int ntiles /* nrows/16 */, int total /* 16*ntiles*32 */) {
  int i = blockIdx.x * 256 + threadIdx.x;
  if (i >= total) return;
  int lane = i & 31;
  int frag = i >> 5;
  int ntile = frag % ntiles;
  int ks = frag / ntiles;
  int n = ntile * 16 + (lane & 15);
  int kbase = ks * 32 + ((lane >= 16) ? 8 : 0);
  const float* src = W + (size_t)n * JJ + kbase;
  v8h lo, hi;
#pragma unroll
  for (int e = 0; e < 8; ++e) { lo[e] = (_Float16)src[e]; hi[e] = (_Float16)src[e + 16]; }
  v8h* dst = (v8h*)(Wp + ((size_t)i << 4));
  dst[0] = lo;
  dst[1] = hi;
}

// =====================================================================
// WMMA projection: Y[r,j] = dot(X[r,:], W[j,:]) (+bias[j]), output f16.
// X: (rows,512) fp32. Wp: packed fragments (ntiles = 32). Y: (rows,512) f16.
// Tile: 32(M) x 512(N), 8 waves x 4 N-tiles x 2 M-subtiles. grid = rows/32.
// =====================================================================
__global__ __launch_bounds__(256) void proj_wmma(
    const float* __restrict__ X, const _Float16* __restrict__ Wp,
    const float* __restrict__ bias, _Float16* __restrict__ Y) {
  __shared__ _Float16 sX[32 * LDS_STRIDE];
  const int tid = threadIdx.x;
  const int m0 = blockIdx.x * 32;

  // Stage X tile (fp32 -> f16) into LDS.
  for (int i = tid; i < 32 * 128; i += 256) {
    int row = i >> 7;
    int k = (i & 127) << 2;
    const float4 x = *(const float4*)(X + (((size_t)(m0 + row)) << 9) + k);
    _Float16* p = sX + row * LDS_STRIDE + k;
    p[0] = (_Float16)x.x; p[1] = (_Float16)x.y;
    p[2] = (_Float16)x.z; p[3] = (_Float16)x.w;
  }
  __syncthreads();

  const int wave = tid >> 5;
  const int lane = tid & 31;
  const int hi = lane >> 4;
  const int col = lane & 15;

  v8f acc[2][4] = {};
#pragma unroll 4
  for (int ks = 0; ks < 16; ++ks) {
    union { v16h v; v8h h[2]; } a[2];
    const int k0 = ks * 32 + (hi << 3);
#pragma unroll
    for (int ms = 0; ms < 2; ++ms) {
      const _Float16* p = sX + (ms * 16 + col) * LDS_STRIDE + k0;
      a[ms].h[0] = *(const v8h*)(p);
      a[ms].h[1] = *(const v8h*)(p + 16);
    }
#pragma unroll
    for (int nt = 0; nt < 4; ++nt) {
      const int ntile = wave * 4 + nt;
      const v16h bf = *(const v16h*)(Wp + ((((size_t)ks * 32 + ntile) * 32 + lane) << 4));
#pragma unroll
      for (int ms = 0; ms < 2; ++ms)
        acc[ms][nt] = __builtin_amdgcn_wmma_f32_16x16x32_f16(
            false, a[ms].v, false, bf, (short)0, acc[ms][nt], false, false);
    }
  }

#pragma unroll
  for (int nt = 0; nt < 4; ++nt) {
    const int n = (wave * 4 + nt) * 16 + col;
    const float b = bias ? bias[n] : 0.f;
#pragma unroll
    for (int ms = 0; ms < 2; ++ms) {
      const int rbase = m0 + ms * 16 + (hi << 3);
#pragma unroll
      for (int r = 0; r < 8; ++r)
        Y[(size_t)(rbase + r) * JJ + n] = (_Float16)(acc[ms][nt][r] + b);
    }
  }
}

// =====================================================================
// Main joint GEMM with fused tanh.
// Workgroup: 256 threads (8 waves), tile 64(M) x 512(N), K=512.
// grid = (MTOT/64, 1024/512) = (2500, 2). Wave: 4 M-subtiles x 4 N-tiles.
// =====================================================================
__global__ __launch_bounds__(256) void joint_main(
    const _Float16* __restrict__ encp,  // (B*T, 512) f16
    const _Float16* __restrict__ decp,  // (B*U, 512) f16
    const _Float16* __restrict__ Wp,    // packed B fragments (ntiles = 64)
    const float* __restrict__ bout,     // (1024)
    float* __restrict__ out)            // (160000, 1024)
{
  __shared__ _Float16 sJ[64 * LDS_STRIDE];   // 66.6 KB joint tile, f16

  const int tid = threadIdx.x;
  const int m0 = blockIdx.x * 64;

  // ---- Stage joint = tanh(enc_proj + dec_proj) tile into LDS (f16) ----
  for (int i = tid; i < 64 * 64; i += 256) {
    int row = i >> 6;
    int k = (i & 63) << 3;
    int m = m0 + row;
    int b = m / TU;
    int r = m - b * TU;
    int t = r / UU;
    int u = r - t * UU;
    const v8h ev = *(const v8h*)(encp + (((size_t)(b * TT + t)) << 9) + k);
    const v8h dv = *(const v8h*)(decp + (((size_t)(b * UU + u)) << 9) + k);
    v8h jv;
#pragma unroll
    for (int q = 0; q < 8; ++q)
      jv[q] = (_Float16)tanhf((float)ev[q] + (float)dv[q]);
    *(v8h*)(sJ + row * LDS_STRIDE + k) = jv;
  }
  __syncthreads();

  const int wave = tid >> 5;
  const int lane = tid & 31;
  const int hi = lane >> 4;     // 0: lanes 0-15, 1: lanes 16-31
  const int col = lane & 15;

  const int nt0 = blockIdx.y * 32 + wave * 4;   // global n-tile base (4 tiles)
  v8f acc[4][4] = {};

#pragma unroll 2
  for (int ks = 0; ks < 16; ++ks) {
    // A fragments (16x32 f16): lane<16 holds row=col, elems 0..7 = K k0..k0+7,
    // elems 8..15 = K k0+16..k0+23; lanes 16..31 shifted by +8 in K.
    union { v16h v; v8h h[2]; } a[4];
    const int k0 = ks * 32 + (hi << 3);
#pragma unroll
    for (int ms = 0; ms < 4; ++ms) {
      const _Float16* p = sJ + (ms * 16 + col) * LDS_STRIDE + k0;
      a[ms].h[0] = *(const v8h*)(p);        // ds_load_b128
      a[ms].h[1] = *(const v8h*)(p + 16);   // ds_load_b128
    }
#pragma unroll
    for (int nt = 0; nt < 4; ++nt) {
      const v16h bf =
          *(const v16h*)(Wp + ((((size_t)ks * 64 + nt0 + nt) * 32 + lane) << 4));
#pragma unroll
      for (int ms = 0; ms < 4; ++ms) {
        acc[ms][nt] = __builtin_amdgcn_wmma_f32_16x16x32_f16(
            false, a[ms].v, false, bf, (short)0, acc[ms][nt], false, false);
      }
    }
  }

  // ---- Store C with bias, non-temporal (655 MB streamed once; keep L2 for
  //      the resident operands: packed weights + f16 projections).
#pragma unroll
  for (int nt = 0; nt < 4; ++nt) {
    const int n = (nt0 + nt) * 16 + col;
    const float bias = bout[n];
#pragma unroll
    for (int ms = 0; ms < 4; ++ms) {
      const int rbase = m0 + ms * 16 + (hi << 3);
#pragma unroll
      for (int r = 0; r < 8; ++r) {
        __builtin_nontemporal_store(acc[ms][nt][r] + bias,
                                    out + (size_t)(rbase + r) * VV + n);
      }
    }
  }
}

// =====================================================================
// Launcher
// =====================================================================
extern "C" void kernel_launch(void* const* d_in, const int* in_sizes, int n_in,
                              void* d_out, int out_size, void* d_ws, size_t ws_size,
                              hipStream_t stream) {
  const float* enc  = (const float*)d_in[0];  // (8,200,1,512)
  const float* dec  = (const float*)d_in[1];  // (8,1,100,512)
  const float* Wenc = (const float*)d_in[2];  // (512,512)
  const float* benc = (const float*)d_in[3];  // (512)
  const float* Wdec = (const float*)d_in[4];  // (512,512)
  const float* Wout = (const float*)d_in[5];  // (1024,512)
  const float* bout = (const float*)d_in[6];  // (1024)
  float* out = (float*)d_out;

  char* ws = (char*)d_ws;
  _Float16* encp  = (_Float16*)(ws);                 // 1600*512*2 = 1,638,400 B
  _Float16* decp  = (_Float16*)(ws + 1638400);       //  800*512*2 =   819,200 B
  _Float16* WpOut = (_Float16*)(ws + 2457600);       // 1024*512*2 = 1,048,576 B
  _Float16* WpEnc = (_Float16*)(ws + 3506176);       //  512*512*2 =   524,288 B
  _Float16* WpDec = (_Float16*)(ws + 4030464);       //  512*512*2 =   524,288 B

  // Pack all three weight matrices into WMMA B-fragment layout.
  pack_w<<<(16 * 64 * 32) / 256, 256, 0, stream>>>(Wout, WpOut, 64, 16 * 64 * 32);
  pack_w<<<(16 * 32 * 32) / 256, 256, 0, stream>>>(Wenc, WpEnc, 32, 16 * 32 * 32);
  pack_w<<<(16 * 32 * 32) / 256, 256, 0, stream>>>(Wdec, WpDec, 32, 16 * 32 * 32);

  // WMMA projections (f16 outputs). enc: 1600 rows (+bias); dec: 800 rows.
  proj_wmma<<<1600 / 32, 256, 0, stream>>>(enc, WpEnc, benc, encp);
  proj_wmma<<< 800 / 32, 256, 0, stream>>>(dec, WpDec, nullptr, decp);

  // Main fused tanh + GEMM: 64x512 tiles.
  dim3 grid(MTOT / 64, VV / 512);
  joint_main<<<grid, 256, 0, stream>>>(encp, decp, WpOut, bout, out);
}
